// RGLRU_17394617549235
// MI455X (gfx1250) — compile-verified, matrix operand order checked by reading, not run on hardware
//
#include <hip/hip_runtime.h>

typedef _Float16 h8  __attribute__((ext_vector_type(8)));
typedef _Float16 h16 __attribute__((ext_vector_type(16)));
typedef float    f8  __attribute__((ext_vector_type(8)));

#define BDIM 8
#define TDIM 4096
#define DDIM 1024
#define MTOT (BDIM * TDIM)     // 32768 rows
#define CHUNK 64               // scan chunk length
#define NCHUNK (TDIM / CHUNK)  // 64 chunks
#define NCHAN (BDIM * DDIM)    // 8192 channels

// ---------------------------------------------------------------------------
// f32 -> f16 conversion, 8 elements/iter (2x b128 load -> 1x b128 store)
// ---------------------------------------------------------------------------
__global__ void rglru_cvt_f16x8(const float4* __restrict__ src,
                                h8* __restrict__ dst, int n8) {
  int i = blockIdx.x * blockDim.x + threadIdx.x;
  const int stride = gridDim.x * blockDim.x;
  for (; i < n8; i += stride) {
    const float4 a = src[2 * i];
    const float4 b = src[2 * i + 1];
    h8 o;
    o[0] = (_Float16)a.x; o[1] = (_Float16)a.y;
    o[2] = (_Float16)a.z; o[3] = (_Float16)a.w;
    o[4] = (_Float16)b.x; o[5] = (_Float16)b.y;
    o[6] = (_Float16)b.z; o[7] = (_Float16)b.w;
    dst[i] = o;
  }
}

// ---------------------------------------------------------------------------
// Fused dual NT-GEMM (r/i logits) + gate epilogue, software-pipelined.
//  Xh:[M,K] f16, Wah/Wxh:[N,K] f16 (K contiguous both sides -> NT gemm).
//  rl = Xh@Wah^T, il = Xh@Wxh^T via v_wmma_f32_16x16x32_f16, then
//  a = sigmoid(loglam)^(8*sigmoid(rl+ba)), gx = sqrt(max(1-a^2,1e-6))*sigmoid(il+bx)*x
//  Tile 128x128, BK=32, 8 waves (4Mx2N), 2x4 16x16 tiles per wave per gemm.
// ---------------------------------------------------------------------------
__launch_bounds__(256, 1)
__global__ void rglru_gemm_gate(const _Float16* __restrict__ Xh,
                                const _Float16* __restrict__ Wah,
                                const _Float16* __restrict__ Wxh,
                                const float* __restrict__ Xf,
                                const float* __restrict__ ba,
                                const float* __restrict__ bx,
                                const float* __restrict__ loglam,
                                float* __restrict__ Aout,   // ws: a  [M,N]
                                float* __restrict__ GXout)  // out: gx [M,N]
{
  constexpr int K = DDIM;
  // 40-half row pitch (80B = 20 banks) -> spread b128 LDS accesses over banks
  __shared__ _Float16 sX[128][40];
  __shared__ _Float16 sA[128][40];
  __shared__ _Float16 sB[128][40];

  const int tid  = threadIdx.x;
  const int lane = tid & 31;
  const int wave = tid >> 5;     // 0..7
  const int wm   = wave >> 1;    // 0..3 (M)
  const int wn   = wave & 1;     // 0..1 (N)

  const int n0 = blockIdx.x * 128;
  const int m0 = blockIdx.y * 128;

  f8 accA[2][4], accX[2][4];
  const f8 fz = {};
#pragma unroll
  for (int mt = 0; mt < 2; ++mt)
#pragma unroll
    for (int nt = 0; nt < 4; ++nt) { accA[mt][nt] = fz; accX[mt][nt] = fz; }

  const int lrow = tid >> 1;           // 0..127 (tile row this thread fills)
  const int lseg = (tid & 1) * 16;     // 0 or 16 (k sub-segment)
  const int kh   = (lane >> 4) << 3;   // per-lane K sub-offset: 0 or 8

  const _Float16* gX = Xh  + (size_t)(m0 + lrow) * K + lseg;
  const _Float16* gA = Wah + (size_t)(n0 + lrow) * K + lseg;
  const _Float16* gB = Wxh + (size_t)(n0 + lrow) * K + lseg;

  // stage tile k0=0 into registers
  h8 rx0, rx1, ra0, ra1, rb0, rb1;
  rx0 = *(const h8*)gX;       rx1 = *(const h8*)(gX + 8);
  ra0 = *(const h8*)gA;       ra1 = *(const h8*)(gA + 8);
  rb0 = *(const h8*)gB;       rb1 = *(const h8*)(gB + 8);

  for (int k0 = 0; k0 < K; k0 += 32) {
    __syncthreads();   // previous compute finished reading LDS
    *(h8*)&sX[lrow][lseg]     = rx0;
    *(h8*)&sX[lrow][lseg + 8] = rx1;
    *(h8*)&sA[lrow][lseg]     = ra0;
    *(h8*)&sA[lrow][lseg + 8] = ra1;
    *(h8*)&sB[lrow][lseg]     = rb0;
    *(h8*)&sB[lrow][lseg + 8] = rb1;
    __syncthreads();

    // prefetch next tile into registers: overlaps with WMMA below
    if (k0 + 32 < K) {
      const int ko = k0 + 32;
      rx0 = *(const h8*)(gX + ko);  rx1 = *(const h8*)(gX + ko + 8);
      ra0 = *(const h8*)(gA + ko);  ra1 = *(const h8*)(gA + ko + 8);
      rb0 = *(const h8*)(gB + ko);  rb1 = *(const h8*)(gB + ko + 8);
    }

    // A fragments: row = wm*32 + mt*16 + lane%16; per-lane K pattern per ISA
    h16 af[2];
#pragma unroll
    for (int mt = 0; mt < 2; ++mt) {
      const int r = wm * 32 + mt * 16 + (lane & 15);
      h8 lo = *(const h8*)&sX[r][kh];
      h8 hi = *(const h8*)&sX[r][kh + 16];
      af[mt] = __builtin_shufflevector(lo, hi, 0, 1, 2, 3, 4, 5, 6, 7,
                                                8, 9, 10, 11, 12, 13, 14, 15);
    }
    // B fragments: column n = weight row n (NT gemm) -> same pattern
    h16 baf[4], bxf[4];
#pragma unroll
    for (int nt = 0; nt < 4; ++nt) {
      const int r = wn * 64 + nt * 16 + (lane & 15);
      {
        h8 lo = *(const h8*)&sA[r][kh];
        h8 hi = *(const h8*)&sA[r][kh + 16];
        baf[nt] = __builtin_shufflevector(lo, hi, 0, 1, 2, 3, 4, 5, 6, 7,
                                                   8, 9, 10, 11, 12, 13, 14, 15);
      }
      {
        h8 lo = *(const h8*)&sB[r][kh];
        h8 hi = *(const h8*)&sB[r][kh + 16];
        bxf[nt] = __builtin_shufflevector(lo, hi, 0, 1, 2, 3, 4, 5, 6, 7,
                                                   8, 9, 10, 11, 12, 13, 14, 15);
      }
    }

#pragma unroll
    for (int mt = 0; mt < 2; ++mt)
#pragma unroll
      for (int nt = 0; nt < 4; ++nt) {
        accA[mt][nt] = __builtin_amdgcn_wmma_f32_16x16x32_f16(
            false, af[mt], false, baf[nt], (short)0, accA[mt][nt], false, false);
        accX[mt][nt] = __builtin_amdgcn_wmma_f32_16x16x32_f16(
            false, af[mt], false, bxf[nt], (short)0, accX[mt][nt], false, false);
      }
  }

  // Epilogue: C/D layout -> VGPR j holds (M = j + 8*(lane/16), N = lane%16)
  const int lm = (lane >> 4) << 3;
  const int ln = lane & 15;
#pragma unroll
  for (int nt = 0; nt < 4; ++nt) {
    const int n = n0 + wn * 64 + nt * 16 + ln;
    const float bav = ba[n];
    const float bxv = bx[n];
    const float ll  = loglam[n];
    const float lsig = -__logf(1.0f + __expf(-ll));  // log(sigmoid(loglam))
#pragma unroll
    for (int mt = 0; mt < 2; ++mt) {
      const int mbase = m0 + wm * 32 + mt * 16 + lm;
#pragma unroll
      for (int j = 0; j < 8; ++j) {
        const int m = mbase + j;
        const float rl = accA[mt][nt][j] + bav;
        const float il = accX[mt][nt][j] + bxv;
        const float r  = 1.0f / (1.0f + __expf(-rl));
        const float iv = 1.0f / (1.0f + __expf(-il));
        const float a  = __expf(8.0f * r * lsig);     // sigmoid(ll)^(8r)
        const float gate = sqrtf(fmaxf((1.0f - a) * (1.0f + a), 1e-6f));
        const size_t idx = (size_t)m * DDIM + n;
        Aout[idx]  = a;
        GXout[idx] = gate * iv * Xf[idx];
      }
    }
  }
}

// ---------------------------------------------------------------------------
// Chunked parallel scan over T (linear recurrence h' = a*h + gx).
// Phase A: per (b, chunk, d): P = prod(a), S = local scan with h=0.
// Phase B: per channel: sequential combine over 64 chunks -> chunk entry h.
// Phase C: per (b, chunk, d): replay chunk with correct entry h, write out.
// Index g = b*(NCHUNK*DDIM) + j*DDIM + d  (d fastest -> coalesced).
// ---------------------------------------------------------------------------
__global__ void rglru_scan_partial(const float* __restrict__ A,
                                   const float* __restrict__ GX,
                                   float* __restrict__ P,
                                   float* __restrict__ S) {
  const int g = blockIdx.x * blockDim.x + threadIdx.x;  // 0..NCHAN*NCHUNK-1
  const int d = g & (DDIM - 1);
  const int j = (g >> 10) & (NCHUNK - 1);
  const int b = g >> 16;
  const size_t base = ((size_t)b * TDIM + (size_t)j * CHUNK) * DDIM + d;
  float p = 1.0f, s = 0.0f;
#pragma unroll 4
  for (int t = 0; t < CHUNK; ++t) {
    const size_t o = base + (size_t)t * DDIM;
    const float av = A[o];
    s = fmaf(av, s, GX[o]);
    p *= av;
  }
  P[g] = p;
  S[g] = s;
}

__global__ void rglru_scan_combine(const float* __restrict__ P,
                                   const float* __restrict__ S,
                                   const float* __restrict__ h0,
                                   float* __restrict__ Hs,   // chunk entry states
                                   float* __restrict__ hfin) {
  const int c = blockIdx.x * blockDim.x + threadIdx.x;  // 0..NCHAN-1
  const int b = c >> 10;
  const int d = c & (DDIM - 1);
  float h = h0[c];
#pragma unroll 8
  for (int j = 0; j < NCHUNK; ++j) {
    const int g = (b * NCHUNK + j) * DDIM + d;
    Hs[g] = h;
    h = fmaf(P[g], h, S[g]);
  }
  hfin[c] = h;
}

__global__ void rglru_scan_apply(const float* __restrict__ A,
                                 const float* __restrict__ Hs,
                                 float* __restrict__ out) {  // gx in, h out
  const int g = blockIdx.x * blockDim.x + threadIdx.x;
  const int d = g & (DDIM - 1);
  const int j = (g >> 10) & (NCHUNK - 1);
  const int b = g >> 16;
  const size_t base = ((size_t)b * TDIM + (size_t)j * CHUNK) * DDIM + d;
  float h = Hs[g];
#pragma unroll 4
  for (int t = 0; t < CHUNK; ++t) {
    const size_t o = base + (size_t)t * DDIM;
    h = fmaf(A[o], h, out[o]);
    out[o] = h;
  }
}

// ---------------------------------------------------------------------------
extern "C" void kernel_launch(void* const* d_in, const int* in_sizes, int n_in,
                              void* d_out, int out_size, void* d_ws, size_t ws_size,
                              hipStream_t stream) {
  const float* x   = (const float*)d_in[0];  // (B,T,D)
  const float* h0  = (const float*)d_in[1];  // (B,D)
  const float* Wa  = (const float*)d_in[2];  // (D,D)
  const float* ba  = (const float*)d_in[3];  // (D)
  const float* Wx  = (const float*)d_in[4];  // (D,D)
  const float* bx  = (const float*)d_in[5];  // (D)
  const float* ll  = (const float*)d_in[6];  // (D)

  float* out  = (float*)d_out;                       // (B,T,D) outputs
  float* hfin = out + (size_t)MTOT * DDIM;           // (B,D) h_final

  // ws layout: x_h(64MB) | Wa_h(2MB) | Wx_h(2MB) | a(128MB) | P,S,Hs(3x2MB)
  _Float16* xh  = (_Float16*)d_ws;
  _Float16* wah = xh + (size_t)MTOT * DDIM;
  _Float16* wxh = wah + (size_t)DDIM * DDIM;
  float*    aw  = (float*)(wxh + (size_t)DDIM * DDIM);
  float*    Pw  = aw + (size_t)MTOT * DDIM;
  float*    Sw  = Pw + (size_t)NCHAN * NCHUNK;
  float*    Hsw = Sw + (size_t)NCHAN * NCHUNK;

  rglru_cvt_f16x8<<<2048, 256, 0, stream>>>((const float4*)x, (h8*)xh,
                                            (MTOT * DDIM) / 8);
  rglru_cvt_f16x8<<<256, 256, 0, stream>>>((const float4*)Wa, (h8*)wah,
                                           (DDIM * DDIM) / 8);
  rglru_cvt_f16x8<<<256, 256, 0, stream>>>((const float4*)Wx, (h8*)wxh,
                                           (DDIM * DDIM) / 8);

  dim3 grid(DDIM / 128, MTOT / 128);  // (8, 256)
  rglru_gemm_gate<<<grid, 256, 0, stream>>>(xh, wah, wxh, x, ba, bx, ll, aw, out);

  const int nchunkthreads = NCHAN * NCHUNK;  // 524288
  rglru_scan_partial<<<nchunkthreads / 256, 256, 0, stream>>>(aw, out, Pw, Sw);
  rglru_scan_combine<<<NCHAN / 256, 256, 0, stream>>>(Pw, Sw, h0, Hsw, hfin);
  rglru_scan_apply<<<nchunkthreads / 256, 256, 0, stream>>>(aw, Hsw, out);
}
